// ConvMod3d_15315853377718
// MI455X (gfx1250) — matrix operand order repaired and checked by reading, not compiled, not run on hardware
//
#include <hip/hip_runtime.h>
#include <hip/hip_bf16.h>

typedef __attribute__((ext_vector_type(8)))  _Float16 v8h;
typedef __attribute__((ext_vector_type(16))) _Float16 v16h;
typedef __attribute__((ext_vector_type(8)))  float    v8f;

#define N_SMP  4
#define CIN    64
#define COUT   64
#define SDIM   512
#define DHW    48
#define ODHW   46
#define NTAP   27
#define KTOT   (NTAP * CIN)     // 1728
#define NCHUNK (KTOT / 32)      // 54 K-chunks of 32
#define WPAD   50               // 48 input cols + 2 zero pad (tail subtile + kw)
#define CINP   72               // padded cin stride: 144B = 36 banks -> conflict free
#define HQ     4                // output h rows per block
#define HROWS  6                // input h rows per block (HQ + 2)
#define NROW   (3 * HROWS)      // 18 LDS rows: [kd(3)][hh(6)]
#define LDS_BYTES (NROW * WPAD * CINP * 2)   // 129,600 B (dynamic LDS; WGP has 320KB)

// ---------------------------------------------------------------------------
// 1) style[n][cin] = s[n] . style_weight[cin] + style_bias[cin]      (4x64)
// ---------------------------------------------------------------------------
__global__ void style_kernel(const float* __restrict__ s,
                             const float* __restrict__ sw,
                             const float* __restrict__ sb,
                             float* __restrict__ style) {
    const int t   = threadIdx.x;          // 256 threads = (n, cin)
    const int n   = t >> 6;
    const int cin = t & 63;
    float acc = sb[cin];
    for (int k = 0; k < SDIM; ++k)
        acc += s[n * SDIM + k] * sw[cin * SDIM + k];
    style[n * CIN + cin] = acc;
}

// ---------------------------------------------------------------------------
// 2) demod[n][cout] = rsqrt( sum_{cin,tap} (weight*style)^2 + eps )  (4x64)
// ---------------------------------------------------------------------------
__global__ void demod_kernel(const float* __restrict__ weight,
                             const float* __restrict__ style,
                             float* __restrict__ demod) {
    const int t    = threadIdx.x;         // 256 threads = (n, cout)
    const int n    = t >> 6;
    const int cout = t & 63;
    float sum = 0.f;
    for (int cin = 0; cin < CIN; ++cin) {
        const float st = style[n * CIN + cin];
        const float* wp = weight + (size_t)(cout * CIN + cin) * NTAP;
        for (int tap = 0; tap < NTAP; ++tap) {
            const float v = wp[tap] * st;
            sum += v * v;
        }
    }
    demod[n * COUT + cout] = rsqrtf(sum + 1e-8f);
}

// ---------------------------------------------------------------------------
// 3) Pack modulated+demodulated weights as f16 directly in the WMMA A-fragment
//    layout used here (16-bit A 16x32, contiguous k order, same as the B loads):
//      K = tap*64 + cin ; chunk = tap*2 + cin/32 ; ko = cin%32
//      lane = (cout%16) + 16*(ko>>4)
//      f    = ko&15
//    Amat[n][strip=cout/16][chunk][lane][f]   (4*4*54*32*16 f16 = 864 KB)
// ---------------------------------------------------------------------------
__global__ void pack_kernel(const float* __restrict__ weight,
                            const float* __restrict__ style,
                            const float* __restrict__ demod,
                            _Float16* __restrict__ Amat) {
    const int idx  = blockIdx.x * 256 + threadIdx.x;   // exactly 4*64*1728
    const int n    = idx / (COUT * KTOT);
    const int r    = idx % (COUT * KTOT);
    const int cout = r / KTOT;
    const int r2   = r % KTOT;
    const int cin  = r2 / NTAP;
    const int tap  = r2 % NTAP;

    const float v = weight[(size_t)(cout * CIN + cin) * NTAP + tap]
                  * style[n * CIN + cin] * demod[n * COUT + cout];

    const int ko    = cin & 31;
    const int chunk = tap * 2 + (cin >> 5);
    const int strip = cout >> 4;
    const int m     = cout & 15;
    const int lane  = m + ((ko >> 4) << 4);
    const int f     = ko & 15;

    const size_t off = (((size_t)(n * 4 + strip) * NCHUNK + chunk) * 32 + lane) * 16 + f;
    Amat[off] = (_Float16)v;
}

// ---------------------------------------------------------------------------
// 4) Implicit-GEMM conv. Block = (n, d, h-quad), 128 threads = 4 waves.
//    Wave wv owns output row h0+wv (48 cols, 3 N-subtiles) x ALL 4 Cout strips:
//    each B fragment from LDS feeds 4 WMMAs (LDS bandwidth / 4), each A
//    fragment from L2 feeds 3 WMMAs. 12 v_wmma per K-chunk per wave,
//    2592 per block, 96 accumulator VGPRs per lane.
// ---------------------------------------------------------------------------
__global__ __launch_bounds__(128)
void conv_wmma_kernel(const float* __restrict__ x,
                      const _Float16* __restrict__ Amat,
                      const float* __restrict__ bias,
                      float* __restrict__ y) {
    extern __shared__ __align__(16) _Float16 xs[];   // [NROW][WPAD][CINP] = 129.6 KB

    const int h0  = blockIdx.x * HQ;
    const int d   = blockIdx.y;
    const int n   = blockIdx.z;
    const int tid = threadIdx.x;

    // ---- stage input patch to LDS as f16: rowIdx = kd*6 + hh ----
    // Consecutive threads walk w: coalesced global reads; LDS store stride
    // 144B = 36 banks -> conflict-free across 16 lanes.
    for (int e = tid; e < CIN * NROW * WPAD; e += 128) {
        const int cin = e / (NROW * WPAD);
        const int rw  = e % (NROW * WPAD);
        const int row = rw / WPAD;
        const int w   = rw % WPAD;
        const int kd  = row / HROWS;
        const int hh  = row % HROWS;
        const int hin = h0 + hh;
        float v = 0.f;
        if (w < DHW && hin < DHW) {
            v = x[(((size_t)(n * CIN + cin) * DHW + (d + kd)) * DHW + hin) * DHW + w];
        }
        xs[(row * WPAD + w) * CINP + cin] = (_Float16)v;
    }
    __syncthreads();

    const int lane  = tid & 31;
    const int wv    = tid >> 5;        // wave id -> output row h0+wv
    const int laneN = lane & 15;       // WMMA N column
    const int laneH = lane >> 4;       // lane half: K sub-range (B) / M+8 (C)

    v8f acc[4][3];
#pragma unroll
    for (int st = 0; st < 4; ++st)
#pragma unroll
        for (int ws = 0; ws < 3; ++ws) acc[st][ws] = v8f{};

    // A fragment base: index = ((n*4+strip)*54 + chunk)*32 + lane (v16h units)
    const v16h* Abase = (const v16h*)Amat + (size_t)n * 4 * NCHUNK * 32 + lane;

    for (int chunk = 0; chunk < NCHUNK; ++chunk) {
        const int tap = chunk >> 1;
        const int kd  = tap / 9;
        const int kh  = (tap / 3) % 3;
        const int kw  = tap % 3;
        const int row = kd * HROWS + wv + kh;
        const int cinOff = ((chunk & 1) << 5) + (laneH << 4);  // 16 contiguous cin

        const _Float16* bp = &xs[(row * WPAD + kw + laneN) * CINP + cinOff];
        v16h b[3];
#pragma unroll
        for (int ws = 0; ws < 3; ++ws) {
            const _Float16* p = bp + ws * 16 * CINP;           // w += 16 per subtile
            const v8h b0 = *(const v8h*)(p);
            const v8h b1 = *(const v8h*)(p + 8);
            b[ws] = __builtin_shufflevector(b0, b1,
                0, 1, 2, 3, 4, 5, 6, 7, 8, 9, 10, 11, 12, 13, 14, 15);
        }

#pragma unroll
        for (int st = 0; st < 4; ++st) {
            const v16h a = Abase[((size_t)st * NCHUNK + chunk) * 32];
#pragma unroll
            for (int ws = 0; ws < 3; ++ws) {
                acc[st][ws] = __builtin_amdgcn_wmma_f32_16x16x32_f16(
                    false, a, false, b[ws], (short)0, acc[st][ws], false, false);
            }
        }
    }

    // ---- epilogue: bias + store (C/D layout: lane = N col, VGPR r = M row) ----
    const int h = h0 + wv;
    if (h < ODHW) {
#pragma unroll
        for (int st = 0; st < 4; ++st) {
#pragma unroll
            for (int ws = 0; ws < 3; ++ws) {
                const int w = ws * 16 + laneN;
                if (w < ODHW) {
#pragma unroll
                    for (int r = 0; r < 8; ++r) {
                        const int cout = (st << 4) + (laneH << 3) + r;
                        y[((((size_t)n * COUT + cout) * ODHW + d) * ODHW + h) * ODHW + w]
                            = acc[st][ws][r] + bias[cout];
                    }
                }
            }
        }
    }
}

// ---------------------------------------------------------------------------
extern "C" void kernel_launch(void* const* d_in, const int* in_sizes, int n_in,
                              void* d_out, int out_size, void* d_ws, size_t ws_size,
                              hipStream_t stream) {
    const float* x      = (const float*)d_in[0];
    const float* s      = (const float*)d_in[1];
    const float* sw     = (const float*)d_in[2];
    const float* sb     = (const float*)d_in[3];
    const float* weight = (const float*)d_in[4];
    const float* bias   = (const float*)d_in[5];
    float* out = (float*)d_out;

    // workspace layout: style (1KB) | demod (1KB) | Amat (864KB) -> ~0.87 MB
    float*    style = (float*)d_ws;
    float*    demod = style + 256;
    _Float16* Amat  = (_Float16*)((char*)d_ws + 2048);

    style_kernel<<<1, 256, 0, stream>>>(s, sw, sb, style);
    demod_kernel<<<1, 256, 0, stream>>>(weight, style, demod);
    pack_kernel<<<(N_SMP * COUT * KTOT) / 256, 256, 0, stream>>>(weight, style, demod, Amat);

    dim3 grid((ODHW + HQ - 1) / HQ, ODHW, N_SMP);   // 12 x 46 x 4 blocks
    conv_wmma_kernel<<<grid, 128, LDS_BYTES, stream>>>(x, Amat, bias, out);
}
